// MultiHeadSelfAttention_23794118819990
// MI455X (gfx1250) — compile-verified, matrix operand order checked by reading, not run on hardware
//
#include <hip/hip_runtime.h>
#include <hip/hip_bf16.h>

typedef __attribute__((ext_vector_type(16))) _Float16 v16h;
typedef __attribute__((ext_vector_type(8)))  float    v8f;
typedef __attribute__((ext_vector_type(2)))  _Float16 h2;

#define B_   4
#define N_   2048
#define C_   768
#define H_   12
#define HD_  64
#define C3_  2304

__device__ __forceinline__ int lane_id() { return threadIdx.x & 31; }

__device__ __forceinline__ v8f wmma32(v16h a, v16h b, v8f c) {
  // D = A(16x32 f16) * B(32x16 f16) + C(16x16 f32)
  return __builtin_amdgcn_wmma_f32_16x16x32_f16(false, a, false, b, (short)0, c, false, false);
}

// A-fragment (16x32, f16) from f32 row-major source. ISA layout:
// lane m (0-15): row m, K = {0..7, 16..23}; lane m+16: row m, K = {8..15, 24..31}
__device__ __forceinline__ v16h load_a_f32(const float* __restrict__ p, int ld, int row0, int k0) {
  const int l = lane_id(), m = l & 15, g = l >> 4;
  const float* r = p + (size_t)(row0 + m) * ld + k0 + 8 * g;
  v16h a;
#pragma unroll
  for (int j = 0; j < 8; ++j) {
    const int kk = ((j & 3) << 1) + ((j >> 2) << 4);
    const float2 t = *reinterpret_cast<const float2*>(r + kk);
    a[2 * j] = (_Float16)t.x;
    a[2 * j + 1] = (_Float16)t.y;
  }
  return a;
}

// A-fragment (16x32) from f16 row-major source (also serves as B-fragment when
// the source is the column-major view of B, i.e. lane indexes the output column).
__device__ __forceinline__ v16h load_a_f16(const _Float16* __restrict__ p, int ld, int row0, int k0) {
  const int l = lane_id(), m = l & 15, g = l >> 4;
  const _Float16* r = p + (size_t)(row0 + m) * ld + k0 + 8 * g;
  v16h a;
#pragma unroll
  for (int j = 0; j < 8; ++j) {
    const int kk = ((j & 3) << 1) + ((j >> 2) << 4);
    const h2 t = *reinterpret_cast<const h2*>(r + kk);
    a[2 * j] = t.x;
    a[2 * j + 1] = t.y;
  }
  return a;
}

// ---------------- Kernel 1: qkv = x @ qkv_w^T  (f32 in, f16 out) ----------------
// M = B*N = 8192, K = 768, Nc = 2304. One wave computes a 32x64 tile.
__global__ __launch_bounds__(256) void gemm_qkv(const float* __restrict__ X,
                                                const float* __restrict__ W,
                                                _Float16* __restrict__ Out) {
  const int wid = threadIdx.x >> 5;
  const int job = blockIdx.x * 8 + wid;
  const int NT = C3_ / 64;               // 36
  const int MT = (B_ * N_) / 32;         // 256
  if (job >= MT * NT) return;
  const int mt = job / NT, nt = job % NT;
  const int row0 = mt * 32, col0 = nt * 64;

  v8f acc[2][4] = {};
  for (int k0 = 0; k0 < C_; k0 += 32) {
    const v16h a0 = load_a_f32(X, C_, row0, k0);
    const v16h a1 = load_a_f32(X, C_, row0 + 16, k0);
#pragma unroll
    for (int t = 0; t < 4; ++t) {
      const v16h bt = load_a_f32(W, C_, col0 + 16 * t, k0);  // W row-major [2304x768]
      acc[0][t] = wmma32(a0, bt, acc[0][t]);
      acc[1][t] = wmma32(a1, bt, acc[1][t]);
    }
  }
  const int l = lane_id(), cb = l & 15, rb = (l >> 4) * 8;
#pragma unroll
  for (int hh = 0; hh < 2; ++hh)
#pragma unroll
    for (int t = 0; t < 4; ++t)
#pragma unroll
      for (int vr = 0; vr < 8; ++vr)
        Out[(size_t)(row0 + 16 * hh + rb + vr) * C3_ + col0 + 16 * t + cb] =
            (_Float16)acc[hh][t][vr];
}

// ---------------- Kernel 2: in-place RoPE on q and k halves ----------------
__global__ __launch_bounds__(256) void rope_kernel(_Float16* __restrict__ qkv) {
  int idx = blockIdx.x * 256 + threadIdx.x;
  const int total = B_ * N_ * 2 * H_ * 32;
  if (idx >= total) return;
  const int j = idx & 31; int t = idx >> 5;
  const int h = t % H_; t /= H_;
  const int s = t % 2;  t /= 2;
  const int n = t % N_; const int b = t / N_;
  const size_t base = ((size_t)(b * N_ + n)) * C3_ + s * C_ + h * HD_;
  // inv_freq[j] = 10000^(-j/32) = exp(-j * ln(10000)/32)
  const float theta = (float)n * __expf(-(float)j * 0.28782313662425576f);
  float cs, sn;
  __sincosf(theta, &cs, &sn);
  const float x0 = (float)qkv[base + j];
  const float x1 = (float)qkv[base + j + 32];
  qkv[base + j]      = (_Float16)(x0 * cs - x1 * sn);
  qkv[base + j + 32] = (_Float16)(x1 * cs + x0 * sn);
}

// ---------------- Kernel 2b: transpose V into [B,H,hd,N] ----------------
// Makes the P*V B-fragment load contiguous along the key axis (L2-resident).
__global__ __launch_bounds__(256) void transpose_v(const _Float16* __restrict__ qkv,
                                                   _Float16* __restrict__ vT) {
  int idx = blockIdx.x * 256 + threadIdx.x;
  const int total = B_ * H_ * HD_ * N_;
  if (idx >= total) return;
  const int n = idx & (N_ - 1); int t = idx >> 11;
  const int e = t & (HD_ - 1);  t >>= 6;
  const int h = t % H_;         const int b = t / H_;
  vT[idx] = qkv[((size_t)b * N_ + n) * C3_ + 2 * C_ + h * HD_ + e];
}

// ---------------- Kernel 3: causal flash attention ----------------
// One wave handles one (b, h, 16-row q tile). Online softmax over 32-key chunks.
__global__ __launch_bounds__(256) void attn_kernel(const _Float16* __restrict__ qkv,
                                                   const _Float16* __restrict__ vT,
                                                   _Float16* __restrict__ aout) {
  __shared__ _Float16 lds_p[8][16 * 32];  // per-wave P staging (C-layout -> A-layout)
  const int wid = threadIdx.x >> 5, l = lane_id();
  const int job = blockIdx.x * 8 + wid;
  const int QT = N_ / 16;  // 128
  if (job >= B_ * H_ * QT) return;
  const int qt = job % QT; int t = job / QT;
  const int h = t % H_;    const int b = t / H_;
  const int q0 = qt * 16;

  const _Float16* qb  = qkv + ((size_t)b * N_) * C3_ + h * HD_;
  const _Float16* kb  = qb + C_;
  const _Float16* vTb = vT + ((size_t)(b * H_ + h) * HD_) * N_;  // [hd, N]

  v16h qf0 = load_a_f16(qb, C3_, q0, 0);
  v16h qf1 = load_a_f16(qb, C3_, q0, 32);
#pragma unroll
  for (int i = 0; i < 16; ++i) {  // fold scale = hd^-0.5 = 0.125 into Q (exact in f16)
    qf0[i] = (_Float16)((float)qf0[i] * 0.125f);
    qf1[i] = (_Float16)((float)qf1[i] * 0.125f);
  }

  v8f o0 = {}, o1 = {}, o2 = {}, o3 = {};
  float mrow[8], lrow[8];
#pragma unroll
  for (int vr = 0; vr < 8; ++vr) { mrow[vr] = -1e30f; lrow[vr] = 0.f; }
  const int rb = (l >> 4) * 8, cb = l & 15;
  _Float16* pb = &lds_p[wid][0];

  for (int kt = 0; kt <= q0 + 15; kt += 32) {
    // S = Q K^T for keys [kt, kt+32)
    v8f s0 = {}, s1 = {};
    s0 = wmma32(qf0, load_a_f16(kb, C3_, kt, 0), s0);
    s0 = wmma32(qf1, load_a_f16(kb, C3_, kt, 32), s0);
    s1 = wmma32(qf0, load_a_f16(kb, C3_, kt + 16, 0), s1);
    s1 = wmma32(qf1, load_a_f16(kb, C3_, kt + 16, 32), s1);

    if (kt + 31 > q0) {  // diagonal chunk: causal mask
#pragma unroll
      for (int vr = 0; vr < 8; ++vr) {
        const int row = q0 + rb + vr;
        if (kt + cb > row) s0[vr] = -1e30f;
        if (kt + 16 + cb > row) s1[vr] = -1e30f;
      }
    }

    // online softmax (rows live across 16 lanes -> width-16 xor reductions)
    float p0[8], p1[8];
#pragma unroll
    for (int vr = 0; vr < 8; ++vr) {
      float rm = fmaxf(s0[vr], s1[vr]);
#pragma unroll
      for (int off = 8; off >= 1; off >>= 1) rm = fmaxf(rm, __shfl_xor(rm, off, 16));
      const float nm = fmaxf(mrow[vr], rm);
      const float corr = __expf(mrow[vr] - nm);
      const float e0 = __expf(s0[vr] - nm);
      const float e1 = __expf(s1[vr] - nm);
      float rs = e0 + e1;
#pragma unroll
      for (int off = 8; off >= 1; off >>= 1) rs += __shfl_xor(rs, off, 16);
      lrow[vr] = lrow[vr] * corr + rs;
      mrow[vr] = nm;
      o0[vr] *= corr; o1[vr] *= corr; o2[vr] *= corr; o3[vr] *= corr;
      p0[vr] = e0; p1[vr] = e1;
    }

    // C-layout P -> LDS (row-major 16x32 f16) -> reload in A-layout
#pragma unroll
    for (int vr = 0; vr < 8; ++vr) {
      pb[(rb + vr) * 32 + cb]      = (_Float16)p0[vr];
      pb[(rb + vr) * 32 + 16 + cb] = (_Float16)p1[vr];
    }
    asm volatile("s_wait_dscnt 0" ::: "memory");
    const v16h pa = load_a_f16(pb, 32, 0, 0);
    // O += P (16x32) * V (32x64); V fragments read contiguously from V^T
    o0 = wmma32(pa, load_a_f16(vTb, N_, 0, kt), o0);
    o1 = wmma32(pa, load_a_f16(vTb, N_, 16, kt), o1);
    o2 = wmma32(pa, load_a_f16(vTb, N_, 32, kt), o2);
    o3 = wmma32(pa, load_a_f16(vTb, N_, 48, kt), o3);
    asm volatile("s_wait_dscnt 0" ::: "memory");
  }

  // normalize + write [B, N, H*hd] in f16 for the projection GEMM
#pragma unroll
  for (int vr = 0; vr < 8; ++vr) {
    const float inv = 1.0f / lrow[vr];
    const size_t orow = ((size_t)b * N_ + q0 + rb + vr) * C_ + h * HD_;
    aout[orow + cb]      = (_Float16)(o0[vr] * inv);
    aout[orow + 16 + cb] = (_Float16)(o1[vr] * inv);
    aout[orow + 32 + cb] = (_Float16)(o2[vr] * inv);
    aout[orow + 48 + cb] = (_Float16)(o3[vr] * inv);
  }
}

// ---------------- Kernel 4: out = attn_out @ proj_w^T + proj_b (f32 out) ----------------
__global__ __launch_bounds__(256) void gemm_proj(const _Float16* __restrict__ A,
                                                 const float* __restrict__ W,
                                                 const float* __restrict__ bias,
                                                 float* __restrict__ Out) {
  const int wid = threadIdx.x >> 5;
  const int job = blockIdx.x * 8 + wid;
  const int NT = C_ / 64;          // 12
  const int MT = (B_ * N_) / 32;   // 256
  if (job >= MT * NT) return;
  const int mt = job / NT, nt = job % NT;
  const int row0 = mt * 32, col0 = nt * 64;

  v8f acc[2][4] = {};
  for (int k0 = 0; k0 < C_; k0 += 32) {
    const v16h a0 = load_a_f16(A, C_, row0, k0);
    const v16h a1 = load_a_f16(A, C_, row0 + 16, k0);
#pragma unroll
    for (int t = 0; t < 4; ++t) {
      const v16h bt = load_a_f32(W, C_, col0 + 16 * t, k0);
      acc[0][t] = wmma32(a0, bt, acc[0][t]);
      acc[1][t] = wmma32(a1, bt, acc[1][t]);
    }
  }
  const int l = lane_id(), cb = l & 15, rb = (l >> 4) * 8;
#pragma unroll
  for (int hh = 0; hh < 2; ++hh)
#pragma unroll
    for (int t = 0; t < 4; ++t) {
      const float bv = bias[col0 + 16 * t + cb];
#pragma unroll
      for (int vr = 0; vr < 8; ++vr)
        Out[(size_t)(row0 + 16 * hh + rb + vr) * C_ + col0 + 16 * t + cb] =
            acc[hh][t][vr] + bv;
    }
}

extern "C" void kernel_launch(void* const* d_in, const int* in_sizes, int n_in,
                              void* d_out, int out_size, void* d_ws, size_t ws_size,
                              hipStream_t stream) {
  const float* x      = (const float*)d_in[0];
  const float* qkv_w  = (const float*)d_in[1];
  const float* proj_w = (const float*)d_in[2];
  const float* proj_b = (const float*)d_in[3];
  float* out = (float*)d_out;

  _Float16* qkv16  = (_Float16*)d_ws;                        // [B,N,3C] f16: 37.75 MB
  _Float16* aout16 = qkv16 + (size_t)B_ * N_ * C3_;          // [B,N,C]  f16: 12.6 MB
  _Float16* vT     = aout16 + (size_t)B_ * N_ * C_;          // [B,H,hd,N] f16: 12.6 MB

  // 1) QKV projection: 9216 wave-jobs, 8 waves/block
  gemm_qkv<<<1152, 256, 0, stream>>>(x, qkv_w, qkv16);
  // 2) RoPE in place on q,k: 6.29M threads
  rope_kernel<<<24576, 256, 0, stream>>>(qkv16);
  // 2b) V transpose for contiguous P*V fragment loads
  transpose_v<<<24576, 256, 0, stream>>>(qkv16, vT);
  // 3) Flash attention: 6144 wave-jobs
  attn_kernel<<<768, 256, 0, stream>>>(qkv16, vT, aout16);
  // 4) Output projection: 3072 wave-jobs
  gemm_proj<<<384, 256, 0, stream>>>(aout16, proj_w, proj_b, out);
}
